// TRMSparse_Inner_53687091200285
// MI455X (gfx1250) — matrix operand order repaired and checked by reading, not compiled, not run on hardware
//
#include <hip/hip_runtime.h>
#include <stdint.h>

// ---------------- problem constants ----------------
#define BB      2
#define LL      2048
#define DD      2048
#define HH      16
#define KVHH    16
#define DHH     128
#define NH_TOT  (HH + 2*KVHH)     // 48 heads in qkv
#define QKV_N   (NH_TOT * DHH)    // 6144
#define MTOT    (BB*LL)           // 4096

// ---------------- WMMA types ----------------
typedef __attribute__((ext_vector_type(16))) __bf16 v16bf;
typedef __attribute__((ext_vector_type(8)))  float  v8f;

union FragAB { v16bf v; uint32_t u[8]; uint16_t s[16]; };

// ---------------- async global->LDS (CDNA5), guarded ----------------
#if defined(__has_builtin)
#  if __has_builtin(__builtin_amdgcn_global_load_async_to_lds_b128)
#    define HAVE_ASYNC_LDS 1
#  endif
#endif
#ifndef HAVE_ASYNC_LDS
#  define HAVE_ASYNC_LDS 0
#endif

#if HAVE_ASYNC_LDS
typedef __attribute__((__vector_size__(4 * sizeof(int)))) int v4i;
typedef __attribute__((address_space(3))) v4i* lds_v4i_ptr;

__device__ inline void async_copy_b128(const void* g, void* l) {
  // low 32 bits of a generic shared-aperture address are the LDS offset
  __builtin_amdgcn_global_load_async_to_lds_b128(
      (v4i*)(uintptr_t)g,
      (lds_v4i_ptr)(uint32_t)(uintptr_t)l,
      0, 0);
}
__device__ inline void async_wait0() {
  asm volatile("s_wait_asynccnt 0" ::: "memory");
}
#endif

__device__ inline v8f zero8() {
  v8f z;
  #pragma unroll
  for (int i = 0; i < 8; ++i) z[i] = 0.0f;
  return z;
}

__device__ inline uint16_t f2bf(float f) {
  uint32_t x = __float_as_uint(f);
  uint32_t r = x + 0x7FFFu + ((x >> 16) & 1u);   // round-to-nearest-even
  return (uint16_t)(r >> 16);
}
__device__ inline float bf2f(uint16_t h) {
  return __uint_as_float(((uint32_t)h) << 16);
}

// A-fragment K offset for VGPR v, half lh  (16-bit A 16x32 layout, ISA 7.12.2)
__device__ inline int a_koff(int v, int lh) {
  return (v < 4 ? 2*v : 16 + 2*(v-4)) + lh*8;
}
// B-fragment K offset for VGPR v, half lh  (same K order as A fragment)
__device__ inline int b_koff(int v, int lh) { return (v < 4 ? 2*v : 16 + 2*(v-4)) + lh*8; }

// ---------------- kernel 1: f32 -> bf16 convert ----------------
__global__ void cvt_bf16_kernel(const float* __restrict__ src,
                                uint16_t* __restrict__ dst, int n) {
  int i = blockIdx.x * blockDim.x + threadIdx.x;
  int stride = gridDim.x * blockDim.x;
  for (; i < n; i += stride) dst[i] = f2bf(src[i]);
}

// ---------------- kernel 2: bf16 GEMM, bf16 output (QKV projection) -------
// C[M,N] = A[M,K] * B[K,N]; workgroup tile 128x128, 8 waves of 32x64.
__global__ __launch_bounds__(256)
void gemm_bf16_bf16out(const uint16_t* __restrict__ A,
                       const uint16_t* __restrict__ Bw,
                       uint16_t* __restrict__ C,
                       int M, int N, int K) {
  __shared__ uint16_t As[128 * 34];   // [m][k], padded stride
  __shared__ uint16_t Bt[128 * 34];   // [n][k], transposed, padded stride

  const int t    = threadIdx.x;
  const int lane = t & 31;
  const int wv   = t >> 5;       // 0..7
  const int wm   = wv & 3;       // 4 M-strips of 32
  const int wn   = wv >> 2;      // 2 N-strips of 64
  const int lm   = lane & 15;
  const int lh   = lane >> 4;
  const int m0   = blockIdx.y * 128;
  const int n0   = blockIdx.x * 128;

  v8f acc[2][4];
  #pragma unroll
  for (int mf = 0; mf < 2; ++mf)
    #pragma unroll
    for (int nf = 0; nf < 4; ++nf) acc[mf][nf] = zero8();

  const int arow = t >> 1, aseg = (t & 1) * 16;     // A: 2 thr/row, 16 elems
  const int brow = t >> 3, bseg = (t & 7) * 16;     // B: 8 thr/row, 16 elems

  for (int k0 = 0; k0 < K; k0 += 32) {
    if (k0 + 32 < K) {  // prefetch next tiles -> global_prefetch_b8
      __builtin_prefetch(A  + (size_t)(m0 + arow) * K + k0 + 32 + aseg, 0, 1);
      __builtin_prefetch(Bw + (size_t)(k0 + 32 + brow) * N + n0 + bseg, 0, 1);
    }
    __syncthreads();
    { // stage A tile 128x32
      const uint16_t* gp = A + (size_t)(m0 + arow) * K + k0 + aseg;
      uint16_t* lp = As + arow * 34 + aseg;
#if HAVE_ASYNC_LDS
      async_copy_b128(gp,     lp);
      async_copy_b128(gp + 8, lp + 8);
#else
      uint4 d0 = ((const uint4*)gp)[0], d1 = ((const uint4*)gp)[1];
      uint32_t* lw = (uint32_t*)lp;
      lw[0]=d0.x; lw[1]=d0.y; lw[2]=d0.z; lw[3]=d0.w;
      lw[4]=d1.x; lw[5]=d1.y; lw[6]=d1.z; lw[7]=d1.w;
#endif
    }
    { // stage B tile 32x128, transposed into Bt[n][k]
      const uint4* gp = (const uint4*)(Bw + (size_t)(k0 + brow) * N + n0 + bseg);
      uint4 d0 = gp[0], d1 = gp[1];
      uint32_t w[8] = {d0.x,d0.y,d0.z,d0.w,d1.x,d1.y,d1.z,d1.w};
      #pragma unroll
      for (int j = 0; j < 8; ++j) {
        Bt[(bseg + 2*j    ) * 34 + brow] = (uint16_t)(w[j] & 0xffffu);
        Bt[(bseg + 2*j + 1) * 34 + brow] = (uint16_t)(w[j] >> 16);
      }
    }
#if HAVE_ASYNC_LDS
    async_wait0();
#endif
    __syncthreads();

    FragAB a[2], b[4];
    #pragma unroll
    for (int mf = 0; mf < 2; ++mf)
      #pragma unroll
      for (int v = 0; v < 8; ++v)
        a[mf].u[v] = *(const uint32_t*)(As + (wm*32 + mf*16 + lm) * 34 + a_koff(v, lh));
    #pragma unroll
    for (int nf = 0; nf < 4; ++nf)
      #pragma unroll
      for (int v = 0; v < 8; ++v)
        b[nf].u[v] = *(const uint32_t*)(Bt + (wn*64 + nf*16 + lm) * 34 + b_koff(v, lh));

    #pragma unroll
    for (int mf = 0; mf < 2; ++mf)
      #pragma unroll
      for (int nf = 0; nf < 4; ++nf)
        acc[mf][nf] = __builtin_amdgcn_wmma_f32_16x16x32_bf16(
            false, a[mf].v, false, b[nf].v, (short)0, acc[mf][nf], false, false);
  }

  #pragma unroll
  for (int mf = 0; mf < 2; ++mf)
    #pragma unroll
    for (int nf = 0; nf < 4; ++nf)
      #pragma unroll
      for (int r = 0; r < 8; ++r) {
        int row = m0 + wm*32 + mf*16 + r + 8*lh;
        int col = n0 + wn*64 + nf*16 + lm;
        C[(size_t)row * N + col] = f2bf(acc[mf][nf][r]);
      }
}

// ---------------- kernel 3: RoPE in-place on q,k slices of bf16 qkv -------
__global__ void rope_kernel(uint16_t* __restrict__ qkv,
                            const float* __restrict__ cosp,
                            const float* __restrict__ sinp) {
  long i = (long)blockIdx.x * blockDim.x + threadIdx.x;   // B*L*32*64 threads
  int j  = (int)(i & 63); i >>= 6;
  int hd = (int)(i & 31); i >>= 5;        // heads 0..31 = q(16) + k(16)
  int l  = (int)(i & (LL - 1)); i >>= 11;
  int b  = (int)i;
  if (b >= BB) return;
  size_t base = (((size_t)(b * LL + l)) * NH_TOT + hd) * DHH;
  float x1 = bf2f(qkv[base + j]);
  float x2 = bf2f(qkv[base + j + 64]);
  float c = cosp[l * DHH + j];
  float s = sinp[l * DHH + j];            // cos/sin[j+64] == cos/sin[j]
  qkv[base + j]      = f2bf(x1 * c - x2 * s);
  qkv[base + j + 64] = f2bf(x2 * c + x1 * s);
}

// ---------------- kernel 4: flash attention -------------------------------
// qkv: [B][L][48][128] bf16 (q RoPE'd in heads 0..15, k in 16..31, v 32..47)
// ctx: [B][L][H*DH] bf16
__global__ __launch_bounds__(256)
void attn_kernel(const uint16_t* __restrict__ qkv, uint16_t* __restrict__ ctx) {
  __shared__ uint16_t Ks[32 * 132];       // K tile  [key][dh]
  __shared__ uint16_t Vt[128 * 34];       // V tile  [dh][key] (transposed)
  __shared__ uint16_t Ps[8 * 16 * 34];    // per-wave P scratch [row][key]

  const int b  = blockIdx.z, h = blockIdx.y;
  const int q0 = blockIdx.x * 128;
  const int t = threadIdx.x, lane = t & 31, wv = t >> 5;
  const int lm = lane & 15, lh = lane >> 4;
  const float scale = 0.08838834764831845f;   // 1/sqrt(128)

  // Q strip (16 rows x 128) in registers as 4 A-fragments
  FragAB qf[4];
  {
    const int qrow = q0 + wv * 16 + lm;
    const uint16_t* qp = qkv + (((size_t)(b * LL + qrow)) * NH_TOT + h) * DHH;
    #pragma unroll
    for (int c = 0; c < 4; ++c)
      #pragma unroll
      for (int v = 0; v < 8; ++v)
        qf[c].u[v] = *(const uint32_t*)(qp + c * 32 + a_koff(v, lh));
  }

  v8f o[8];
  float mrow[8], lrow[8];
  #pragma unroll
  for (int d = 0; d < 8; ++d) o[d] = zero8();
  #pragma unroll
  for (int r = 0; r < 8; ++r) { mrow[r] = -3.0e38f; lrow[r] = 0.0f; }

  const int krow = t >> 3, kseg = (t & 7) * 16;

  for (int kv0 = 0; kv0 < LL; kv0 += 32) {
    if (kv0 + 32 < LL) {  // prefetch next K/V rows
      __builtin_prefetch(qkv +
          (((size_t)(b * LL + kv0 + 32 + krow)) * NH_TOT + HH + h) * DHH + kseg, 0, 1);
      __builtin_prefetch(qkv +
          (((size_t)(b * LL + kv0 + 32 + krow)) * NH_TOT + HH + KVHH + h) * DHH + kseg, 0, 1);
    }
    __syncthreads();
    { // stage K tile (natural layout == Bt for Q*K^T)
      const uint16_t* gp = qkv +
          (((size_t)(b * LL + kv0 + krow)) * NH_TOT + HH + h) * DHH + kseg;
      uint16_t* lp = Ks + krow * 132 + kseg;
#if HAVE_ASYNC_LDS
      async_copy_b128(gp,     lp);
      async_copy_b128(gp + 8, lp + 8);
#else
      uint4 d0 = ((const uint4*)gp)[0], d1 = ((const uint4*)gp)[1];
      uint32_t* lw = (uint32_t*)lp;
      lw[0]=d0.x; lw[1]=d0.y; lw[2]=d0.z; lw[3]=d0.w;
      lw[4]=d1.x; lw[5]=d1.y; lw[6]=d1.z; lw[7]=d1.w;
#endif
    }
    { // stage V tile transposed
      const uint4* gp = (const uint4*)(qkv +
          (((size_t)(b * LL + kv0 + krow)) * NH_TOT + HH + KVHH + h) * DHH + kseg);
      uint4 d0 = gp[0], d1 = gp[1];
      uint32_t w[8] = {d0.x,d0.y,d0.z,d0.w,d1.x,d1.y,d1.z,d1.w};
      #pragma unroll
      for (int j = 0; j < 8; ++j) {
        Vt[(kseg + 2*j    ) * 34 + krow] = (uint16_t)(w[j] & 0xffffu);
        Vt[(kseg + 2*j + 1) * 34 + krow] = (uint16_t)(w[j] >> 16);
      }
    }
#if HAVE_ASYNC_LDS
    async_wait0();
#endif
    __syncthreads();

    // scores S (16 x 32) = Q * K^T
    v8f sfr[2];
    #pragma unroll
    for (int nf = 0; nf < 2; ++nf) {
      sfr[nf] = zero8();
      #pragma unroll
      for (int c = 0; c < 4; ++c) {
        FragAB bf;
        #pragma unroll
        for (int v = 0; v < 8; ++v)
          bf.u[v] = *(const uint32_t*)(Ks + (nf*16 + lm) * 132 + c*32 + b_koff(v, lh));
        sfr[nf] = __builtin_amdgcn_wmma_f32_16x16x32_bf16(
            false, qf[c].v, false, bf.v, (short)0, sfr[nf], false, false);
      }
    }

    // online softmax (row r of this half lives in element r across 16 lanes)
    float alpha[8];
    #pragma unroll
    for (int r = 0; r < 8; ++r) {
      float s0 = sfr[0][r] * scale, s1 = sfr[1][r] * scale;
      float mx = fmaxf(s0, s1);
      mx = fmaxf(mx, __shfl_xor(mx, 1, 32));
      mx = fmaxf(mx, __shfl_xor(mx, 2, 32));
      mx = fmaxf(mx, __shfl_xor(mx, 4, 32));
      mx = fmaxf(mx, __shfl_xor(mx, 8, 32));
      float mnew = fmaxf(mrow[r], mx);
      alpha[r] = expf(mrow[r] - mnew);
      mrow[r] = mnew;
      float p0 = expf(s0 - mnew), p1 = expf(s1 - mnew);
      sfr[0][r] = p0; sfr[1][r] = p1;
      float rs = p0 + p1;
      rs += __shfl_xor(rs, 1, 32);
      rs += __shfl_xor(rs, 2, 32);
      rs += __shfl_xor(rs, 4, 32);
      rs += __shfl_xor(rs, 8, 32);
      lrow[r] = lrow[r] * alpha[r] + rs;
    }
    #pragma unroll
    for (int d = 0; d < 8; ++d)
      #pragma unroll
      for (int r = 0; r < 8; ++r) o[d][r] = o[d][r] * alpha[r];

    // P: C-layout -> bf16 A-layout via per-wave LDS scratch
    uint16_t* myP = Ps + wv * 16 * 34;
    #pragma unroll
    for (int nf = 0; nf < 2; ++nf)
      #pragma unroll
      for (int r = 0; r < 8; ++r)
        myP[(r + 8*lh) * 34 + nf*16 + lm] = f2bf(sfr[nf][r]);
    FragAB pa;
    #pragma unroll
    for (int v = 0; v < 8; ++v)
      pa.u[v] = *(const uint32_t*)(myP + lm * 34 + a_koff(v, lh));

    // O += P * V
    #pragma unroll
    for (int d = 0; d < 8; ++d) {
      FragAB bv;
      #pragma unroll
      for (int v = 0; v < 8; ++v)
        bv.u[v] = *(const uint32_t*)(Vt + (d*16 + lm) * 34 + b_koff(v, lh));
      o[d] = __builtin_amdgcn_wmma_f32_16x16x32_bf16(
          false, pa.v, false, bv.v, (short)0, o[d], false, false);
    }
  }

  // epilogue: O / l -> ctx bf16
  #pragma unroll
  for (int d = 0; d < 8; ++d)
    #pragma unroll
    for (int r = 0; r < 8; ++r) {
      int row = q0 + wv * 16 + r + 8*lh;
      float val = o[d][r] / lrow[r];
      ctx[((size_t)(b * LL + row)) * DD + h * DHH + d*16 + lm] = f2bf(val);
    }
}

// ---------------- kernel 5: bf16 GEMM, f32 output with mask (Wo) ----------
__global__ __launch_bounds__(256)
void gemm_bf16_f32out_mask(const uint16_t* __restrict__ A,
                           const uint16_t* __restrict__ Bw,
                           float* __restrict__ C,
                           const unsigned char* __restrict__ mask,
                           int M, int N, int K) {
  __shared__ uint16_t As[128 * 34];
  __shared__ uint16_t Bt[128 * 34];

  const int t = threadIdx.x, lane = t & 31, wv = t >> 5;
  const int wm = wv & 3, wn = wv >> 2;
  const int lm = lane & 15, lh = lane >> 4;
  const int m0 = blockIdx.y * 128, n0 = blockIdx.x * 128;

  v8f acc[2][4];
  #pragma unroll
  for (int mf = 0; mf < 2; ++mf)
    #pragma unroll
    for (int nf = 0; nf < 4; ++nf) acc[mf][nf] = zero8();

  const int arow = t >> 1, aseg = (t & 1) * 16;
  const int brow = t >> 3, bseg = (t & 7) * 16;

  for (int k0 = 0; k0 < K; k0 += 32) {
    if (k0 + 32 < K) {
      __builtin_prefetch(A  + (size_t)(m0 + arow) * K + k0 + 32 + aseg, 0, 1);
      __builtin_prefetch(Bw + (size_t)(k0 + 32 + brow) * N + n0 + bseg, 0, 1);
    }
    __syncthreads();
    {
      const uint16_t* gp = A + (size_t)(m0 + arow) * K + k0 + aseg;
      uint16_t* lp = As + arow * 34 + aseg;
#if HAVE_ASYNC_LDS
      async_copy_b128(gp,     lp);
      async_copy_b128(gp + 8, lp + 8);
#else
      uint4 d0 = ((const uint4*)gp)[0], d1 = ((const uint4*)gp)[1];
      uint32_t* lw = (uint32_t*)lp;
      lw[0]=d0.x; lw[1]=d0.y; lw[2]=d0.z; lw[3]=d0.w;
      lw[4]=d1.x; lw[5]=d1.y; lw[6]=d1.z; lw[7]=d1.w;
#endif
    }
    {
      const uint4* gp = (const uint4*)(Bw + (size_t)(k0 + brow) * N + n0 + bseg);
      uint4 d0 = gp[0], d1 = gp[1];
      uint32_t w[8] = {d0.x,d0.y,d0.z,d0.w,d1.x,d1.y,d1.z,d1.w};
      #pragma unroll
      for (int j = 0; j < 8; ++j) {
        Bt[(bseg + 2*j    ) * 34 + brow] = (uint16_t)(w[j] & 0xffffu);
        Bt[(bseg + 2*j + 1) * 34 + brow] = (uint16_t)(w[j] >> 16);
      }
    }
#if HAVE_ASYNC_LDS
    async_wait0();
#endif
    __syncthreads();

    FragAB a[2], b[4];
    #pragma unroll
    for (int mf = 0; mf < 2; ++mf)
      #pragma unroll
      for (int v = 0; v < 8; ++v)
        a[mf].u[v] = *(const uint32_t*)(As + (wm*32 + mf*16 + lm) * 34 + a_koff(v, lh));
    #pragma unroll
    for (int nf = 0; nf < 4; ++nf)
      #pragma unroll
      for (int v = 0; v < 8; ++v)
        b[nf].u[v] = *(const uint32_t*)(Bt + (wn*64 + nf*16 + lm) * 34 + b_koff(v, lh));

    #pragma unroll
    for (int mf = 0; mf < 2; ++mf)
      #pragma unroll
      for (int nf = 0; nf < 4; ++nf)
        acc[mf][nf] = __builtin_amdgcn_wmma_f32_16x16x32_bf16(
            false, a[mf].v, false, b[nf].v, (short)0, acc[mf][nf], false, false);
  }

  #pragma unroll
  for (int mf = 0; mf < 2; ++mf)
    #pragma unroll
    for (int r = 0; r < 8; ++r) {
      int row = m0 + wm*32 + mf*16 + r + 8*lh;
      float mv = (float)mask[row];
      #pragma unroll
      for (int nf = 0; nf < 4; ++nf) {
        int col = n0 + wn*64 + nf*16 + lm;
        C[(size_t)row * N + col] = acc[mf][nf][r] * mv;
      }
    }
}

// ---------------- host launcher -------------------------------------------
// inputs: hidden_states f32[B,L,D], cos f32[L,DH], sin f32[L,DH],
//         active_mask bool[B,L], Wqkv f32[D,6144], Wo f32[2048,2048]
// output: f32[B,L,D]
extern "C" void kernel_launch(void* const* d_in, const int* in_sizes, int n_in,
                              void* d_out, int out_size, void* d_ws, size_t ws_size,
                              hipStream_t stream) {
  const float* hidden = (const float*)d_in[0];
  const float* cosp   = (const float*)d_in[1];
  const float* sinp   = (const float*)d_in[2];
  const unsigned char* mask = (const unsigned char*)d_in[3]; // jax bool = 1 byte
  const float* Wqkv   = (const float*)d_in[4];
  const float* Wo     = (const float*)d_in[5];
  float* out = (float*)d_out;

  // workspace layout (bf16 elements), total ~117 MB
  uint16_t* hb    = (uint16_t*)d_ws;                     // 4096*2048
  uint16_t* wqkvb = hb    + (size_t)MTOT * DD;           // 2048*6144
  uint16_t* wob   = wqkvb + (size_t)DD * QKV_N;          // 2048*2048
  uint16_t* qkvb  = wob   + (size_t)DD * DD;             // 4096*6144
  uint16_t* ctxb  = qkvb  + (size_t)MTOT * QKV_N;        // 4096*2048

  cvt_bf16_kernel<<<4096, 256, 0, stream>>>(hidden, hb,    MTOT * DD);
  cvt_bf16_kernel<<<4096, 256, 0, stream>>>(Wqkv,   wqkvb, DD * QKV_N);
  cvt_bf16_kernel<<<4096, 256, 0, stream>>>(Wo,     wob,   DD * DD);

  gemm_bf16_bf16out<<<dim3(QKV_N/128, MTOT/128), 256, 0, stream>>>(
      hb, wqkvb, qkvb, MTOT, QKV_N, DD);

  rope_kernel<<<(BB*LL*32*64)/256, 256, 0, stream>>>(qkvb, cosp, sinp);

  attn_kernel<<<dim3(LL/128, HH, BB), 256, 0, stream>>>(qkvb, ctxb);

  gemm_bf16_f32out_mask<<<dim3(DD/128, MTOT/128), 256, 0, stream>>>(
      ctxb, wob, out, mask, MTOT, DD, DD);
}
